// StochasticDurationPredictor_26121991094747
// MI455X (gfx1250) — matrix operand order repaired and checked by reading, not compiled, not run on hardware
//
#include <hip/hip_runtime.h>
#include <math.h>
#include <stdint.h>

#define EPI_LN   1
#define EPI_MASK 2
#define EPI_RES  4

namespace {

constexpr int Cc  = 192;
constexpr int Kc  = 3;
constexpr int NBc = 10;
constexpr int Pc  = 3 * NBc - 1;   // 29
constexpr int Lc  = 3;
constexpr int Bn  = 32;
constexpr int Tn  = 768;
constexpr int TT  = 64;            // time tile per block in the WMMA kernel
constexpr int CPAD = 200;          // padded K-row length for the LDS input tile
constexpr float TBf = 5.0f;
constexpr float LOG2PIf = 1.8378770664093453f;

typedef __attribute__((ext_vector_type(16))) _Float16 v16h;
typedef __attribute__((ext_vector_type(8)))  _Float16 v8h;
typedef __attribute__((ext_vector_type(8)))  float    v8f;
typedef __attribute__((ext_vector_type(4)))  unsigned int v4u;
typedef __attribute__((ext_vector_type(8)))  int v8i_t;
typedef __attribute__((ext_vector_type(4)))  int v4i_t;

union V16u { v16h v; v8h h[2]; };

__device__ __forceinline__ float gelu_f(float x) {
  float x3 = x * x * x;
  return 0.5f * x * (1.0f + tanhf(0.7978845608028654f * (x + 0.044715f * x3)));
}
__device__ __forceinline__ float logsig_f(float x) {
  return (x < 0.f) ? (x - log1pf(expf(x))) : (-log1pf(expf(-x)));
}
__device__ __forceinline__ float softplus_f(float x) {
  return (x > 20.f) ? x : log1pf(expf(x));
}

// Issue one 1-D TDM descriptor: copy nhalf f16 elements from gsrc to LDS.
__device__ __forceinline__ void tdm_load_f16(const _Float16* gsrc, unsigned lds_off,
                                             unsigned nhalf)
{
  unsigned long long ga = (unsigned long long)(uintptr_t)gsrc;
  v4u g0;
  g0[0] = 1u;                                        // count = 1 (valid user D#)
  g0[1] = lds_off;                                   // LDS byte address
  g0[2] = (unsigned)(ga & 0xffffffffu);              // global_addr[31:0]
  g0[3] = (unsigned)((ga >> 32) & 0x1ffffffu) | (2u << 30);  // addr[56:32] | type=2
  v8i_t g1;
  g1[0] = (1 << 16);                                 // data_size = 1 -> 2 bytes
  g1[1] = (int)((nhalf & 0xffffu) << 16);            // tensor_dim0[15:0]
  g1[2] = (int)((nhalf >> 16) & 0xffffu);            // tensor_dim0[31:16]
  g1[3] = (int)((nhalf & 0xffffu) << 16);            // tile_dim0
  g1[4] = 0;                                         // tile_dim1/2 unused
  g1[5] = (int)nhalf;                                // tensor_dim0_stride low32
  g1[6] = 0;
  g1[7] = 0;
  v4i_t gz; gz[0] = 0; gz[1] = 0; gz[2] = 0; gz[3] = 0;
#if __clang_major__ >= 23
  v8i_t g5;
#pragma unroll
  for (int q = 0; q < 8; ++q) g5[q] = 0;
  __builtin_amdgcn_tensor_load_to_lds(g0, g1, gz, gz, g5, 0);
#else
  __builtin_amdgcn_tensor_load_to_lds(g0, g1, gz, gz, 0);
#endif
}

// ---------------------------------------------------------------------------
// Fused conv1x1 (GEMM over channels) via WMMA f16 -> f32.
// Weights arrive pre-packed as f16 and are staged to LDS by the TDM.
// Optional epilogue: bias + LayerNorm(C) + GELU, residual add, mask.
// Block: waves = ceil(CO/16); each wave computes a 16(co) x 64(t) strip.
// ---------------------------------------------------------------------------
__global__ void __launch_bounds__(384)
k_conv1x1_wmma(const float* __restrict__ X,      // (B, 192, T)
               const _Float16* __restrict__ Wh,  // (CO, 192) packed f16
               const float* __restrict__ bias,   // (CO)
               const float* __restrict__ lng,    // (192) or null
               const float* __restrict__ lnb,    // (192) or null
               const float* __restrict__ res,    // (B, CO, T) or null
               const float* __restrict__ mask,   // (B, 1, T)
               float* __restrict__ Y,            // (B, CO, T)
               int CO, int flags)
{
  __shared__ _Float16 Ws[Cc * Cc];               // staged weights (<=72 KB)
  __shared__ _Float16 Bs[TT * CPAD];             // input tile, t-major (25 KB)
  __shared__ float sSum[TT], sSum2[TT], sMean[TT], sRstd[TT];

  const int b   = blockIdx.y;
  const int t0  = blockIdx.x * TT;
  const int tid = threadIdx.x;
  const int wave   = tid >> 5;
  const int lane   = tid & 31;
  const int laneLo = lane & 15;
  const int laneHi = lane >> 4;

  // TDM: one DMA pulls the whole packed weight matrix into LDS.
  if (tid == 0)
    tdm_load_f16(Wh, (unsigned)(uintptr_t)(void*)Ws, (unsigned)(CO * Cc));

  // Cooperative stage of the K x TT input tile to LDS (fp32 -> f16), t-major.
  for (int i = tid; i < Cc * TT; i += blockDim.x) {
    int kk = i >> 6;          // TT == 64
    int tt = i & (TT - 1);
    Bs[tt * CPAD + kk] = (_Float16)X[((size_t)b * Cc + kk) * Tn + t0 + tt];
  }
  if (flags & EPI_LN) {
    for (int i = tid; i < TT; i += blockDim.x) { sSum[i] = 0.f; sSum2[i] = 0.f; }
  }
  const int m = wave * 16 + laneLo;     // A-matrix row owned by this lane
  if (flags & EPI_RES)
    __builtin_prefetch(res + ((size_t)b * CO + (m < CO ? m : 0)) * Tn + t0, 0, 1);

  if (tid < 32) __builtin_amdgcn_s_wait_tensorcnt(0);  // wave0: DMA done
  __syncthreads();                                     // release everyone

  v8f acc[4];
#pragma unroll
  for (int s = 0; s < 4; ++s)
#pragma unroll
    for (int r = 0; r < 8; ++r) acc[s][r] = 0.f;

#pragma unroll
  for (int ks = 0; ks < 6; ++ks) {              // K = 192 in steps of 32
    // A fragment: lane holds K = laneHi*8 + {0..7} and 16 + laneHi*8 + {0..7}
    const _Float16* arow = Ws + m * Cc + ks * 32 + laneHi * 8;
    V16u ua;
    ua.h[0] = *(const v8h*)(arow);
    ua.h[1] = *(const v8h*)(arow + 16);
#pragma unroll
    for (int s = 0; s < 4; ++s) {
      // B fragment: lane = column t, K = laneHi*8 + {0..7} and 16 + laneHi*8 + {0..7}
      const _Float16* brow = Bs + (s * 16 + laneLo) * CPAD + ks * 32 + laneHi * 8;
      V16u ub;
      ub.h[0] = *(const v8h*)(brow);
      ub.h[1] = *(const v8h*)(brow + 16);
      acc[s] = __builtin_amdgcn_wmma_f32_16x16x32_f16(
          false, ua.v, false, ub.v, (short)0, acc[s], false, false);
    }
  }

  // ---- epilogue ----
  if (flags & EPI_LN) {
    // C/D layout: VGPR r -> channel wave*16 + r + laneHi*8, column t = s*16+laneLo
#pragma unroll
    for (int s = 0; s < 4; ++s) {
      float ps = 0.f, ps2 = 0.f;
#pragma unroll
      for (int r = 0; r < 8; ++r) {
        int c = wave * 16 + r + laneHi * 8;
        float v = acc[s][r] + bias[c];
        ps += v; ps2 += v * v;
      }
      int tt = s * 16 + laneLo;
      atomicAdd(&sSum[tt], ps);
      atomicAdd(&sSum2[tt], ps2);
    }
    __syncthreads();
    for (int i = tid; i < TT; i += blockDim.x) {
      float mn = sSum[i] * (1.f / Cc);
      sMean[i] = mn;
      sRstd[i] = rsqrtf(sSum2[i] * (1.f / Cc) - mn * mn + 1e-5f);
    }
    __syncthreads();
  }

#pragma unroll
  for (int s = 0; s < 4; ++s) {
    int tt = s * 16 + laneLo;
    int tg = t0 + tt;
    float mk = (flags & EPI_MASK) ? mask[(size_t)b * Tn + tg] : 1.f;
#pragma unroll
    for (int r = 0; r < 8; ++r) {
      int c = wave * 16 + r + laneHi * 8;
      if (c < CO) {
        float v = acc[s][r] + bias[c];
        if (flags & EPI_LN) {
          v = (v - sMean[tt]) * sRstd[tt] * lng[c] + lnb[c];
          v = gelu_f(v);
        }
        if (flags & EPI_RES) v += res[((size_t)b * CO + c) * Tn + tg];
        v *= mk;
        Y[((size_t)b * CO + c) * Tn + tg] = v;
      }
    }
  }
}

// ---------------------------------------------------------------------------
// Depthwise dilated conv(K=3) on x*mask, + bias, + LayerNorm over C + GELU.
// Block = 192 threads (one per channel) for one (b, t).
// ---------------------------------------------------------------------------
__global__ void __launch_bounds__(Cc)
k_dw_ln_gelu(const float* __restrict__ x, const float* __restrict__ mask,
             const float* __restrict__ dw, const float* __restrict__ db,
             const float* __restrict__ g, const float* __restrict__ bb,
             int dil, float* __restrict__ y)
{
  __shared__ float sSum, sSum2, sMean, sRstd;
  const int c = threadIdx.x;
  const int t = blockIdx.x;
  const int b = blockIdx.y;

  float a = db[c];
#pragma unroll
  for (int k = 0; k < Kc; ++k) {
    int tt = t + dil * (k - 1);
    if (tt >= 0 && tt < Tn)
      a += x[((size_t)b * Cc + c) * Tn + tt] * mask[(size_t)b * Tn + tt] * dw[c * Kc + k];
  }
  if (c == 0) { sSum = 0.f; sSum2 = 0.f; }
  __syncthreads();
  // wave32 shuffle tree, then one LDS atomic per wave
  float s1 = a, s2v = a * a;
#pragma unroll
  for (int o = 16; o > 0; o >>= 1) {
    s1  += __shfl_down(s1, o);
    s2v += __shfl_down(s2v, o);
  }
  if ((c & 31) == 0) { atomicAdd(&sSum, s1); atomicAdd(&sSum2, s2v); }
  __syncthreads();
  if (c == 0) {
    float mn = sSum * (1.f / Cc);
    sMean = mn;
    sRstd = rsqrtf(sSum2 * (1.f / Cc) - mn * mn + 1e-5f);
  }
  __syncthreads();
  float v = (a - sMean) * sRstd * g[c] + bb[c];
  y[((size_t)b * Cc + c) * Tn + t] = gelu_f(v);
}

// fp32 -> f16 weight packing
__global__ void k_cvt_f16(const float* __restrict__ s, _Float16* __restrict__ d, int n)
{
  int i = blockIdx.x * blockDim.x + threadIdx.x;
  if (i < n) d[i] = (_Float16)s[i];
}

// 1->C channel expansion (pre conv with CI=1), fused with optional +g conditioning
__global__ void k_expand(const float* __restrict__ src, int srcBStride,
                         const float* __restrict__ wv, const float* __restrict__ bv,
                         const float* __restrict__ g, float* __restrict__ y)
{
  int i = blockIdx.x * blockDim.x + threadIdx.x;
  int b = blockIdx.y;
  if (i >= Cc * Tn) return;
  int c = i / Tn;
  int t = i - c * Tn;
  float v = wv[c] * src[(size_t)b * srcBStride + t] + bv[c];
  if (g) v += g[((size_t)b * Cc + c) * Tn + t];
  y[((size_t)b * Cc + c) * Tn + t] = v;
}

__global__ void k_add_inplace(float* __restrict__ a, const float* __restrict__ bsrc, int n)
{
  int i = blockIdx.x * blockDim.x + threadIdx.x;
  if (i < n) a[i] += bsrc[i];
}

__global__ void k_zero(float* __restrict__ p, int n)
{
  int i = blockIdx.x * blockDim.x + threadIdx.x;
  if (i < n) p[i] = 0.f;
}

// elementwise affine on (B,2,T): z' = (m + exp(logs)*z)*mask ; acc += sum(logs)*mask
__global__ void k_affine(const float* __restrict__ z, const float* __restrict__ mask,
                         const float* __restrict__ m2, const float* __restrict__ logs2,
                         float* __restrict__ zout, float* __restrict__ acc, int accIdx)
{
  int t = blockIdx.x * blockDim.x + threadIdx.x;
  int b = blockIdx.y;
  if (t >= Tn) return;
  float msk = mask[(size_t)b * Tn + t];
  float v0 = (m2[0] + expf(logs2[0]) * z[((size_t)b * 2 + 0) * Tn + t]) * msk;
  float v1 = (m2[1] + expf(logs2[1]) * z[((size_t)b * 2 + 1) * Tn + t]) * msk;
  zout[((size_t)b * 2 + 0) * Tn + t] = v0;
  zout[((size_t)b * 2 + 1) * Tn + t] = v1;
  atomicAdd(&acc[b * 4 + accIdx], (logs2[0] + logs2[1]) * msk);
}

// Rational-quadratic spline flow on channel 1, identity on channel 0,
// writes the flipped output, accumulates log|det| into acc[b*4+accIdx].
__global__ void k_rqs(const float* __restrict__ z, const float* __restrict__ hp, // (B,P,T)
                      const float* __restrict__ mask, float* __restrict__ zout,
                      float* __restrict__ acc, int accIdx)
{
  int t = blockIdx.x * blockDim.x + threadIdx.x;
  int b = blockIdx.y;
  if (t >= Tn) return;

  const float mb = 0.001f;
  const float cpad = logf(expm1f(1.0f - mb));
  const float sc = 0.07216878364870323f;         // 1/sqrt(192)

  float msk = mask[(size_t)b * Tn + t];
  float x0 = z[((size_t)b * 2 + 0) * Tn + t];
  float x1 = z[((size_t)b * 2 + 1) * Tn + t];

  float uw[NBc], uh[NBc], cw[NBc + 1], ch[NBc + 1], dv[NBc + 1];
#pragma unroll
  for (int i = 0; i < NBc; ++i) {
    uw[i] = hp[((size_t)b * Pc + i) * Tn + t] * sc;
    uh[i] = hp[((size_t)b * Pc + NBc + i) * Tn + t] * sc;
  }
  // widths
  float mx = uw[0];
#pragma unroll
  for (int i = 1; i < NBc; ++i) mx = fmaxf(mx, uw[i]);
  float se = 0.f;
#pragma unroll
  for (int i = 0; i < NBc; ++i) { uw[i] = expf(uw[i] - mx); se += uw[i]; }
  float inv = 1.f / se, run = 0.f;
  cw[0] = -TBf;
#pragma unroll
  for (int i = 0; i < NBc; ++i) {
    run += mb + (1.f - mb * NBc) * uw[i] * inv;
    cw[i + 1] = 2.f * TBf * run - TBf;
  }
  cw[NBc] = TBf;
  // heights
  mx = uh[0];
#pragma unroll
  for (int i = 1; i < NBc; ++i) mx = fmaxf(mx, uh[i]);
  se = 0.f;
#pragma unroll
  for (int i = 0; i < NBc; ++i) { uh[i] = expf(uh[i] - mx); se += uh[i]; }
  inv = 1.f / se; run = 0.f;
  ch[0] = -TBf;
#pragma unroll
  for (int i = 0; i < NBc; ++i) {
    run += mb + (1.f - mb * NBc) * uh[i] * inv;
    ch[i + 1] = 2.f * TBf * run - TBf;
  }
  ch[NBc] = TBf;
  // derivatives (padded with cpad on both ends)
  dv[0]   = mb + softplus_f(cpad);
  dv[NBc] = mb + softplus_f(cpad);
#pragma unroll
  for (int j = 1; j < NBc; ++j)
    dv[j] = mb + softplus_f(hp[((size_t)b * Pc + 2 * NBc + (j - 1)) * Tn + t]);

  float xc = fminf(fmaxf(x1, -TBf), TBf);
  int idx = -1;
#pragma unroll
  for (int i = 0; i < NBc; ++i) idx += (xc >= cw[i]) ? 1 : 0;
  idx = (idx < 0) ? 0 : ((idx > NBc - 1) ? NBc - 1 : idx);

  float icw = 0.f, iwid = 1.f, ich = 0.f, ihei = 1.f, d0 = 1.f, d1 = 1.f;
#pragma unroll
  for (int i = 0; i < NBc; ++i) {
    bool sel = (i == idx);
    icw  = sel ? cw[i] : icw;
    iwid = sel ? (cw[i + 1] - cw[i]) : iwid;
    ich  = sel ? ch[i] : ich;
    ihei = sel ? (ch[i + 1] - ch[i]) : ihei;
  }
#pragma unroll
  for (int j = 0; j <= NBc; ++j) {
    d0 = (j == idx)     ? dv[j] : d0;
    d1 = (j == idx + 1) ? dv[j] : d1;
  }

  float idelta = ihei / iwid;
  float th  = (xc - icw) / iwid;
  float t1m = th * (1.f - th);
  float den = idelta + (d0 + d1 - 2.f * idelta) * t1m;
  float outv = ich + ihei * (idelta * th * th + d0 * t1m) / den;
  float omt = 1.f - th;
  float dnum = idelta * idelta * (d1 * th * th + 2.f * idelta * t1m + d0 * omt * omt);
  float lad = logf(dnum) - 2.f * logf(den);

  bool inside = (x1 >= -TBf) && (x1 <= TBf);
  float y1 = inside ? outv : x1;
  float ladm = inside ? lad : 0.f;

  // write flipped: new ch0 = x1', new ch1 = x0
  zout[((size_t)b * 2 + 1) * Tn + t] = x0 * msk;
  zout[((size_t)b * 2 + 0) * Tn + t] = y1 * msk;
  atomicAdd(&acc[b * 4 + accIdx], ladm * msk);
}

// zu/u/z0/y0 stage between the two flow stacks
__global__ void k_latent(const float* __restrict__ zq, const float* __restrict__ e,
                         const float* __restrict__ wdur, const float* __restrict__ mask,
                         float* __restrict__ zout, float* __restrict__ acc)
{
  int t = blockIdx.x * blockDim.x + threadIdx.x;
  int b = blockIdx.y;
  if (t >= Tn) return;
  float msk = mask[(size_t)b * Tn + t];
  float zu = zq[((size_t)b * 2 + 0) * Tn + t];
  float z1 = zq[((size_t)b * 2 + 1) * Tn + t];
  float u  = msk / (1.f + expf(-zu));
  float z0 = (wdur[(size_t)b * Tn + t] - u) * msk;
  float e0 = e[((size_t)b * 2 + 0) * Tn + t];
  float e1 = e[((size_t)b * 2 + 1) * Tn + t];
  float gq = -0.5f * (LOG2PIf + e0 * e0) * msk - 0.5f * (LOG2PIf + e1 * e1) * msk;
  float y0 = logf(fmaxf(z0, 1e-5f)) * msk;
  zout[((size_t)b * 2 + 0) * Tn + t] = y0;
  zout[((size_t)b * 2 + 1) * Tn + t] = z1;
  atomicAdd(&acc[b * 4 + 0], (logsig_f(zu) + logsig_f(-zu)) * msk);  // ldq
  atomicAdd(&acc[b * 4 + 1], gq);                                    // gauss(q)
  atomicAdd(&acc[b * 4 + 2], -y0);                                   // ldt
}

__global__ void k_gauss(const float* __restrict__ z, const float* __restrict__ mask,
                        float* __restrict__ acc)
{
  int t = blockIdx.x * blockDim.x + threadIdx.x;
  int b = blockIdx.y;
  if (t >= Tn) return;
  float msk = mask[(size_t)b * Tn + t];
  float z0 = z[((size_t)b * 2 + 0) * Tn + t];
  float z1 = z[((size_t)b * 2 + 1) * Tn + t];
  atomicAdd(&acc[b * 4 + 3],
            0.5f * (LOG2PIf + z0 * z0) * msk + 0.5f * (LOG2PIf + z1 * z1) * msk);
}

__global__ void k_final(const float* __restrict__ acc, float* __restrict__ out)
{
  int b = threadIdx.x;
  if (b < Bn)
    out[b] = (acc[b * 4 + 3] - acc[b * 4 + 2]) + (acc[b * 4 + 1] - acc[b * 4 + 0]);
}

} // namespace

extern "C" void kernel_launch(void* const* d_in, const int* in_sizes, int n_in,
                              void* d_out, int out_size, void* d_ws, size_t ws_size,
                              hipStream_t stream)
{
  (void)in_sizes; (void)n_in; (void)out_size; (void)ws_size;

  const float* x        = (const float*)d_in[0];
  const float* mask     = (const float*)d_in[1];
  const float* wdur     = (const float*)d_in[2];
  const float* e_q      = (const float*)d_in[3];
  const float* pre_w    = (const float*)d_in[4];
  const float* pre_b    = (const float*)d_in[5];
  const float* proj_w   = (const float*)d_in[6];
  const float* proj_b   = (const float*)d_in[7];
  const float* dds_dep_w= (const float*)d_in[8];
  const float* dds_dep_b= (const float*)d_in[9];
  const float* dds_pw_w = (const float*)d_in[10];
  const float* dds_pw_b = (const float*)d_in[11];
  const float* dds_ln_g = (const float*)d_in[12];
  const float* dds_ln_b = (const float*)d_in[13];
  const float* post_pre_w = (const float*)d_in[14];
  const float* post_pre_b = (const float*)d_in[15];
  const float* post_proj_w= (const float*)d_in[16];
  const float* post_proj_b= (const float*)d_in[17];
  const float* pdds_dep_w = (const float*)d_in[18];
  const float* pdds_dep_b = (const float*)d_in[19];
  const float* pdds_pw_w  = (const float*)d_in[20];
  const float* pdds_pw_b  = (const float*)d_in[21];
  const float* pdds_ln_g  = (const float*)d_in[22];
  const float* pdds_ln_b  = (const float*)d_in[23];
  const float* aff_m      = (const float*)d_in[24];
  const float* aff_logs   = (const float*)d_in[25];
  const float* f_pre_w    = (const float*)d_in[26];
  const float* f_pre_b    = (const float*)d_in[27];
  const float* f_dep_w    = (const float*)d_in[28];
  const float* f_dep_b    = (const float*)d_in[29];
  const float* f_pw_w     = (const float*)d_in[30];
  const float* f_pw_b     = (const float*)d_in[31];
  const float* f_ln_g     = (const float*)d_in[32];
  const float* f_ln_b     = (const float*)d_in[33];
  const float* f_proj_w   = (const float*)d_in[34];
  const float* f_proj_b   = (const float*)d_in[35];
  const float* paff_m     = (const float*)d_in[36];
  const float* paff_logs  = (const float*)d_in[37];
  const float* p_pre_w    = (const float*)d_in[38];
  const float* p_pre_b    = (const float*)d_in[39];
  const float* p_dep_w    = (const float*)d_in[40];
  const float* p_dep_b    = (const float*)d_in[41];
  const float* p_pw_w     = (const float*)d_in[42];
  const float* p_pw_b     = (const float*)d_in[43];
  const float* p_ln_g     = (const float*)d_in[44];
  const float* p_ln_b     = (const float*)d_in[45];
  const float* p_proj_w   = (const float*)d_in[46];
  const float* p_proj_b   = (const float*)d_in[47];

  float* out = (float*)d_out;

  // ---- workspace carve ----
  char* ws = (char*)d_ws;
  size_t off = 0;
  auto allocB = [&](size_t bytes) -> void* {
    void* p = (void*)(ws + off);
    off += ((bytes + 255) / 256) * 256;
    return p;
  };
  const size_t NCT = (size_t)Bn * Cc * Tn;
  float* hbuf = (float*)allocB(NCT * 4);           // h (persists)
  float* gp   = (float*)allocB(NCT * 4);           // hw then h+hw (persists)
  float* bufX = (float*)allocB(NCT * 4);           // dds working
  float* bufT = (float*)allocB(NCT * 4);           // depthwise output
  float* bufY = (float*)allocB(NCT * 4);           // conv output / ping-pong
  float* zbuf = (float*)allocB((size_t)Bn * 2 * Tn * 4);
  float* z2   = (float*)allocB((size_t)Bn * 2 * Tn * 4);
  float* hp   = (float*)allocB((size_t)Bn * Pc * Tn * 4);
  float* acc  = (float*)allocB(Bn * 4 * 4);

  // f16-packed weights
  const int MSQ = Cc * Cc;                // 36864
  const int MPJ = Pc * Cc;                // 5568
  _Float16* wh = (_Float16*)allocB((size_t)(33 * MSQ + 8 * MPJ) * 2);
  _Float16* wh_pre     = wh;
  _Float16* wh_proj    = wh + 1 * MSQ;
  _Float16* wh_postproj= wh + 2 * MSQ;
  _Float16* wh_dds     = wh + 3 * MSQ;    // 3 mats
  _Float16* wh_pdds    = wh + 6 * MSQ;    // 3 mats
  _Float16* wh_p_pw    = wh + 9 * MSQ;    // 12 mats
  _Float16* wh_f_pw    = wh + 21 * MSQ;   // 12 mats
  _Float16* wh_p_proj  = wh + 33 * MSQ;            // 4 x (29x192)
  _Float16* wh_f_proj  = wh + 33 * MSQ + 4 * MPJ;  // 4 x (29x192)

  auto cvt = [&](const float* s, _Float16* d, int n) {
    k_cvt_f16<<<(n + 255) / 256, 256, 0, stream>>>(s, d, n);
  };
  cvt(pre_w,       wh_pre,      MSQ);
  cvt(proj_w,      wh_proj,     MSQ);
  cvt(post_proj_w, wh_postproj, MSQ);
  cvt(dds_pw_w,    wh_dds,      3 * MSQ);
  cvt(pdds_pw_w,   wh_pdds,     3 * MSQ);
  cvt(p_pw_w,      wh_p_pw,     12 * MSQ);
  cvt(f_pw_w,      wh_f_pw,     12 * MSQ);
  cvt(p_proj_w,    wh_p_proj,   4 * MPJ);
  cvt(f_proj_w,    wh_f_proj,   4 * MPJ);

  const dim3 gridG(Tn / TT, Bn);           // WMMA GEMM grid
  const dim3 gridDW(Tn, Bn);               // depthwise grid
  const dim3 gridT((Tn + 255) / 256, Bn);  // per-(b,t) elementwise
  const dim3 gridE((Cc * Tn + 255) / 256, Bn);

  auto conv = [&](const float* X_, const _Float16* W_, const float* b_,
                  const float* g_, const float* lb_, const float* r_, float* Y_,
                  int CO, int flags) {
    int waves = (CO + 15) / 16;
    k_conv1x1_wmma<<<gridG, waves * 32, 0, stream>>>(X_, W_, b_, g_, lb_, r_, mask,
                                                     Y_, CO, flags);
  };

  // dds: 3 x (depthwise+LN+gelu -> conv1x1+LN+gelu+residual). returns result ptr.
  auto run_dds = [&](float* xin, const float* depw, const float* depb,
                     const _Float16* pww, const float* pwb,
                     const float* lngp, const float* lnbp) -> float* {
    float* cur = xin;
    float* other = (xin == bufX) ? bufY : bufX;
    int dil = 1;
    for (int i = 0; i < Lc; ++i) {
      k_dw_ln_gelu<<<gridDW, Cc, 0, stream>>>(cur, mask,
          depw + (size_t)i * Cc * Kc, depb + (size_t)i * Cc,
          lngp + (size_t)(i * 2 + 0) * Cc, lnbp + (size_t)(i * 2 + 0) * Cc,
          dil, bufT);
      conv(bufT, pww + (size_t)i * MSQ, pwb + (size_t)i * Cc,
           lngp + (size_t)(i * 2 + 1) * Cc, lnbp + (size_t)(i * 2 + 1) * Cc,
           cur, other, Cc, EPI_LN | EPI_RES);
      float* tmp = cur; cur = other; other = tmp;
      dil *= Kc;
    }
    return cur;
  };

  k_zero<<<1, 128, 0, stream>>>(acc, Bn * 4);

  // ---- h path: pre conv -> dds -> proj conv (*mask) ----
  conv(x, wh_pre, pre_b, nullptr, nullptr, nullptr, bufX, Cc, 0);
  float* px = run_dds(bufX, dds_dep_w, dds_dep_b, wh_dds, dds_pw_b, dds_ln_g, dds_ln_b);
  conv(px, wh_proj, proj_b, nullptr, nullptr, nullptr, hbuf, Cc, EPI_MASK);

  // ---- hw path ----
  k_expand<<<gridE, 256, 0, stream>>>(wdur, Tn, post_pre_w, post_pre_b, nullptr, bufX);
  px = run_dds(bufX, pdds_dep_w, pdds_dep_b, wh_pdds, pdds_pw_b, pdds_ln_g, pdds_ln_b);
  conv(px, wh_postproj, post_proj_b, nullptr, nullptr, nullptr, gp, Cc, EPI_MASK);
  k_add_inplace<<<((int)NCT + 255) / 256, 256, 0, stream>>>(gp, hbuf, (int)NCT);

  // ---- posterior flows (g = gp), accumulate ldq into acc[...,0] ----
  k_affine<<<gridT, 256, 0, stream>>>(e_q, mask, paff_m, paff_logs, zbuf, acc, 0);
  for (int i = 0; i < 4; ++i) {
    k_expand<<<gridE, 256, 0, stream>>>(zbuf, 2 * Tn, p_pre_w + (size_t)i * Cc,
                                        p_pre_b + (size_t)i * Cc, gp, bufX);
    px = run_dds(bufX, p_dep_w + (size_t)i * Lc * Cc * Kc, p_dep_b + (size_t)i * Lc * Cc,
                 wh_p_pw + (size_t)i * Lc * MSQ, p_pw_b + (size_t)i * Lc * Cc,
                 p_ln_g + (size_t)i * Lc * 2 * Cc, p_ln_b + (size_t)i * Lc * 2 * Cc);
    conv(px, wh_p_proj + (size_t)i * MPJ, p_proj_b + (size_t)i * Pc,
         nullptr, nullptr, nullptr, hp, Pc, EPI_MASK);
    k_rqs<<<gridT, 256, 0, stream>>>(zbuf, hp, mask, z2, acc, 0);
    float* tmp = zbuf; zbuf = z2; z2 = tmp;
  }

  // ---- latent bridge + affine ----
  k_latent<<<gridT, 256, 0, stream>>>(zbuf, e_q, wdur, mask, z2, acc);
  k_affine<<<gridT, 256, 0, stream>>>(z2, mask, aff_m, aff_logs, zbuf, acc, 2);

  // ---- forward flows (g = h), accumulate ldt into acc[...,2] ----
  for (int i = 0; i < 4; ++i) {
    k_expand<<<gridE, 256, 0, stream>>>(zbuf, 2 * Tn, f_pre_w + (size_t)i * Cc,
                                        f_pre_b + (size_t)i * Cc, hbuf, bufX);
    px = run_dds(bufX, f_dep_w + (size_t)i * Lc * Cc * Kc, f_dep_b + (size_t)i * Lc * Cc,
                 wh_f_pw + (size_t)i * Lc * MSQ, f_pw_b + (size_t)i * Lc * Cc,
                 f_ln_g + (size_t)i * Lc * 2 * Cc, f_ln_b + (size_t)i * Lc * 2 * Cc);
    conv(px, wh_f_proj + (size_t)i * MPJ, f_proj_b + (size_t)i * Pc,
         nullptr, nullptr, nullptr, hp, Pc, EPI_MASK);
    k_rqs<<<gridT, 256, 0, stream>>>(zbuf, hp, mask, z2, acc, 2);
    float* tmp = zbuf; zbuf = z2; z2 = tmp;
  }

  // ---- final NLL ----
  k_gauss<<<gridT, 256, 0, stream>>>(zbuf, mask, acc);
  k_final<<<1, 32, 0, stream>>>(acc, out);
}
